// NetVLADConsensus_41669772705901
// MI455X (gfx1250) — matrix operand order repaired and checked by reading, not compiled, not run on hardware
//
#include <hip/hip_runtime.h>
#include <hip/hip_bf16.h>

#define NB 32
#define TT 1024
#define CC 512
#define KK 64

typedef __bf16 bf16;
typedef bf16  v16bf __attribute__((ext_vector_type(16)));
typedef float v8f   __attribute__((ext_vector_type(8)));

union ABFrag { v16bf v; uint4 q[2]; };
union BF4    { bf16 v[4]; uint2 u; };

// ---------------- fp32 -> bf16 conversion (W) ----------------
__global__ __launch_bounds__(256)
void cvt_bf16_kernel(const float* __restrict__ src, bf16* __restrict__ dst, int n) {
    int i = blockIdx.x * 256 + threadIdx.x;
    if (i < n) dst[i] = (bf16)src[i];
}

__global__ __launch_bounds__(256)
void zero_kernel(float* __restrict__ p, int n) {
    int i = blockIdx.x * 256 + threadIdx.x;
    if (i < n) p[i] = 0.f;
}

// ---------------- x: fp32 -> bf16, producing BOTH layouts ----------------
// xbf[N][T][C] (t-major, A-operand for GEMM-1) and xT[N][C][T] (c-major,
// B-operand for GEMM-2). One fp32 read, two bf16 writes; 32x32 tile via LDS.
// grid: NB * (TT/32) * (CC/32) = NB*512 blocks, 256 threads.
__global__ __launch_bounds__(256)
void cvtT_kernel(const float* __restrict__ x, bf16* __restrict__ xbf,
                 bf16* __restrict__ xT)
{
    const int bx  = blockIdx.x;
    const int n   = bx >> 9;                 // 512 tiles per n
    const int rem = bx & 511;
    const int t0  = (rem >> 4) * 32;
    const int c0  = (rem & 15) * 32;
    const int tid = threadIdx.x;

    __shared__ bf16 tile[32][36];            // 72 B row stride, 8 B aligned

    {
        const int t  = tid >> 3;
        const int cg = (tid & 7) * 4;
        const float* p = x + ((size_t)n * TT + t0 + t) * CC + c0 + cg;
        BF4 r;
        r.v[0] = (bf16)p[0]; r.v[1] = (bf16)p[1];
        r.v[2] = (bf16)p[2]; r.v[3] = (bf16)p[3];
        *(uint2*)(xbf + ((size_t)n * TT + t0 + t) * CC + c0 + cg) = r.u;
        *(uint2*)&tile[t][cg] = r.u;
    }
    __syncthreads();
    {
        const int c  = tid >> 3;
        const int tg = (tid & 7) * 4;
        BF4 r;
        r.v[0] = tile[tg + 0][c];
        r.v[1] = tile[tg + 1][c];
        r.v[2] = tile[tg + 2][c];
        r.v[3] = tile[tg + 3][c];
        *(uint2*)(xT + ((size_t)n * CC + c0 + c) * TT + t0 + tg) = r.u;
    }
}

// ---------------- pass 1: logits + softmax -> a (bf16), asum ----------------
// grid: NB * (TT/32) blocks, 256 threads (8 waves).
// Each wave computes one 16(t) x 16(k) logits tile via 16 bf16 WMMA steps over C.
__global__ __launch_bounds__(256)
void assign_kernel(const bf16* __restrict__ xbf, const bf16* __restrict__ wbf,
                   const float* __restrict__ bias, bf16* __restrict__ abf,
                   float* __restrict__ asum)
{
    const int n    = blockIdx.x >> 5;          // TT/32 == 32 tiles per n
    const int t0   = (blockIdx.x & 31) * 32;
    const int tid  = threadIdx.x;
    const int lane = tid & 31;
    const int wave = tid >> 5;
    const int ttile = wave >> 2;               // 0..1
    const int ktile = wave & 3;                // 0..3
    const int half  = lane >> 4;
    const int l16   = lane & 15;

    __shared__ float logits[32][KK + 1];
    __shared__ float aw[32][KK + 1];

    // A fragment source: x row t = t0 + ttile*16 + (lane&15)
    const bf16* xr = xbf + ((size_t)n * TT + (t0 + ttile * 16 + l16)) * CC;
    // B fragment source: W row k = ktile*16 + (lane&15)  (B = W^T column == W row)
    const bf16* wr = wbf + (size_t)(ktile * 16 + l16) * CC;

    v8f acc = {};
    for (int c0 = 0; c0 < CC; c0 += 32) {
        ABFrag a, b;
        // A 16x32 bf16 layout: element e -> K = (e/8)*16 + half*8 + (e%8)
        const bf16* pa = xr + c0 + half * 8;
        a.q[0] = *(const uint4*)(pa);
        a.q[1] = *(const uint4*)(pa + 16);
        // B 32x16 bf16 layout: element e -> K = (e/8)*16 + half*8 + (e%8)
        const bf16* pb = wr + c0 + half * 8;
        b.q[0] = *(const uint4*)(pb);
        b.q[1] = *(const uint4*)(pb + 16);
        __builtin_prefetch(pa + 64, 0, 3);     // locality 3 -> WGP-scope prefetch
        acc = __builtin_amdgcn_wmma_f32_16x16x32_bf16(false, a.v, false, b.v,
                                                      (short)0, acc, false, false);
    }

    // D layout: VGPR r -> row m = half*8 + r, col = lane&15
    {
        const float bk = bias[ktile * 16 + l16];
        const int tb = ttile * 16 + half * 8;
#pragma unroll
        for (int r = 0; r < 8; ++r)
            logits[tb + r][ktile * 16 + l16] = acc[r] + bk;
    }
    __syncthreads();

    // softmax over K per t-row (32 rows, one thread each; K=64 from LDS)
    if (tid < 32) {
        float mx = -1e30f;
        for (int k = 0; k < KK; ++k) mx = fmaxf(mx, logits[tid][k]);
        float s = 0.f;
        for (int k = 0; k < KK; ++k) {
            float e = __expf(logits[tid][k] - mx);
            aw[tid][k] = e; s += e;
        }
        float inv = 1.f / s;
        for (int k = 0; k < KK; ++k) aw[tid][k] *= inv;
    }
    __syncthreads();

    // asum[n][k] += sum_t a
    if (tid < KK) {
        float s = 0.f;
        for (int t = 0; t < 32; ++t) s += aw[t][tid];
        atomicAdd(&asum[n * KK + tid], s);
    }

    // store a as bf16 [N][K][T] (t contiguous -> A-frag ready for pass 2)
    {
        const int k  = tid >> 2;
        const int ts = (tid & 3) * 8;
        bf16* dst = abf + ((size_t)n * KK + k) * TT + t0 + ts;
#pragma unroll
        for (int j = 0; j < 8; ++j) dst[j] = (bf16)aw[ts + j][k];
    }
}

// ---------------- pass 2: vlad = a^T x - asum*centroid ----------------
// grid: NB * (CC/32) blocks, 256 threads (8 waves).
// Block tile: K=64 x 32 c-columns; T streamed in steps of 32.
// Both operands are t-contiguous in memory (abf and xT), so the whole loop is
// barrier-free pipelined b128 loads + WMMA, same as pass 1.
__global__ __launch_bounds__(256)
void vlad_kernel(const bf16* __restrict__ xT, const bf16* __restrict__ abf,
                 const float* __restrict__ asum, const float* __restrict__ cent,
                 float* __restrict__ out)
{
    const int n    = blockIdx.x >> 4;          // CC/32 == 16 tiles per n
    const int c0   = (blockIdx.x & 15) * 32;
    const int tid  = threadIdx.x;
    const int lane = tid & 31;
    const int wave = tid >> 5;
    const int ktile = wave & 3;                // 0..3
    const int ctile = wave >> 2;               // 0..1
    const int half  = lane >> 4;
    const int l16   = lane & 15;

    // A = a tile 16(k) x 32(t): row k = ktile*16 + (lane&15)
    const bf16* ar = abf + ((size_t)n * KK + (ktile * 16 + l16)) * TT;
    // B = x^T tile 32(t) x 16(c): column c = c0 + ctile*16 + (lane&15)
    const bf16* br = xT + ((size_t)n * CC + (c0 + ctile * 16 + l16)) * TT;

    v8f acc = {};
    for (int t0 = 0; t0 < TT; t0 += 32) {
        ABFrag a, b;
        // element e -> t = t0 + (e/8)*16 + half*8 + (e%8)
        const bf16* pa = ar + t0 + half * 8;
        a.q[0] = *(const uint4*)(pa);
        a.q[1] = *(const uint4*)(pa + 16);
        // element e -> t = t0 + (e/8)*16 + half*8 + (e%8)
        const bf16* pb = br + t0 + half * 8;
        b.q[0] = *(const uint4*)(pb);
        b.q[1] = *(const uint4*)(pb + 16);
        __builtin_prefetch(pb + 64, 0, 3);
        acc = __builtin_amdgcn_wmma_f32_16x16x32_bf16(false, a.v, false, b.v,
                                                      (short)0, acc, false, false);
    }

    // epilogue: row m -> cluster k, col -> channel c
    const int c = c0 + ctile * 16 + l16;
#pragma unroll
    for (int r = 0; r < 8; ++r) {
        const int k = ktile * 16 + half * 8 + r;
        float v = acc[r] - asum[n * KK + k] * cent[k * CC + c];
        out[((size_t)n * KK + k) * CC + c] = v;
    }
}

// ---------------- pass 3: intra + global L2 normalization ----------------
__global__ __launch_bounds__(256)
void norm_kernel(float* __restrict__ out)
{
    const int n = blockIdx.x;
    const int tid = threadIdx.x;
    float* v = out + (size_t)n * KK * CC;

    __shared__ float red[256];
    float gss = 0.f;

    for (int k = 0; k < KK; ++k) {
        float* row = v + k * CC;
        float ss = 0.f;
        for (int c = tid; c < CC; c += 256) { float x = row[c]; ss += x * x; }
        red[tid] = ss;
        __syncthreads();
        for (int s = 128; s > 0; s >>= 1) {
            if (tid < s) red[tid] += red[tid + s];
            __syncthreads();
        }
        const float rss = red[0];
        __syncthreads();                       // protect red[] before next write
        const float inv = 1.f / fmaxf(sqrtf(rss), 1e-12f);
        for (int c = tid; c < CC; c += 256) row[c] *= inv;
        gss += rss * inv * inv;                // uniform across threads
    }
    __syncthreads();
    const float ginv = 1.f / fmaxf(sqrtf(gss), 1e-12f);
    for (int i = tid; i < KK * CC; i += 256) v[i] *= ginv;
}

// ---------------- launcher ----------------
extern "C" void kernel_launch(void* const* d_in, const int* in_sizes, int n_in,
                              void* d_out, int out_size, void* d_ws, size_t ws_size,
                              hipStream_t stream)
{
    const float* x    = (const float*)d_in[0];   // [N,T,C]
    const float* W    = (const float*)d_in[1];   // [K,C]
    const float* b    = (const float*)d_in[2];   // [K]
    const float* cent = (const float*)d_in[3];   // [K,C]
    float* out = (float*)d_out;                  // [N, K*C]

    char* ws = (char*)d_ws;
    const size_t xbf_bytes = (size_t)NB * TT * CC * sizeof(bf16);   // 32 MB
    const size_t xT_bytes  = (size_t)NB * CC * TT * sizeof(bf16);   // 32 MB
    const size_t wbf_bytes = (size_t)KK * CC * sizeof(bf16);        // 64 KB
    const size_t abf_bytes = (size_t)NB * KK * TT * sizeof(bf16);   // 4 MB
    bf16*  xbf  = (bf16*)(ws);
    bf16*  xT   = (bf16*)(ws + xbf_bytes);
    bf16*  wbf  = (bf16*)(ws + xbf_bytes + xT_bytes);
    bf16*  abf  = (bf16*)(ws + xbf_bytes + xT_bytes + wbf_bytes);
    float* asum = (float*)(ws + xbf_bytes + xT_bytes + wbf_bytes + abf_bytes);

    const int wn = KK * CC;
    cvtT_kernel<<<NB * 512, 256, 0, stream>>>(x, xbf, xT);
    cvt_bf16_kernel<<<(wn + 255) / 256, 256, 0, stream>>>(W, wbf, wn);
    zero_kernel<<<(NB * KK + 255) / 256, 256, 0, stream>>>(asum, NB * KK);

    assign_kernel<<<NB * (TT / 32), 256, 0, stream>>>(xbf, wbf, b, abf, asum);
    vlad_kernel<<<NB * (CC / 32), 256, 0, stream>>>(xT, abf, asum, cent, out);
    norm_kernel<<<NB, 256, 0, stream>>>(out);
}
